// DCNv2_Slow_137438953845
// MI455X (gfx1250) — matrix operand order repaired and checked by reading, not compiled, not run on hardware
//
#include <hip/hip_runtime.h>

typedef __bf16 bf16;
typedef __attribute__((ext_vector_type(16))) __bf16 v16bf;
typedef __attribute__((ext_vector_type(8)))  __bf16 v8bf;
typedef __attribute__((ext_vector_type(8)))  float  v8f;

#define N_    4
#define C_    256
#define H_    64
#define W_    64
#define OC_   256
#define K2_   9
#define HW_   (H_ * W_)          // 4096
#define KDIM  (C_ * K2_)         // 2304
#define NCOL  (N_ * HW_)         // 16384

__device__ __forceinline__ unsigned short f2bf(float f) {
    unsigned int u = __float_as_uint(f);
    unsigned int r = u + 0x7FFFu + ((u >> 16) & 1u);   // round-to-nearest-even
    return (unsigned short)(r >> 16);
}

// -------- Stage 1: 27-channel 3x3 offset conv (direct) --------------------
__global__ void __launch_bounds__(256)
offset_conv_kernel(const float* __restrict__ x, const float* __restrict__ ow,
                   const float* __restrict__ ob, float* __restrict__ om) {
    int idx = blockIdx.x * blockDim.x + threadIdx.x;
    if (idx >= N_ * 27 * HW_) return;
    int wo = idx & 63;
    int ho = (idx >> 6) & 63;
    int oc = (idx >> 12) % 27;
    int n  = (idx >> 12) / 27;

    float acc = ob[oc];
    const float* xb = x + (size_t)n * C_ * HW_;
    const float* wb = ow + (size_t)oc * C_ * 9;
    for (int c = 0; c < C_; ++c) {
        const float* xc = xb + c * HW_;
        const float* wc = wb + c * 9;
#pragma unroll
        for (int ky = 0; ky < 3; ++ky) {
            int y = ho - 1 + ky;
            if (y < 0 || y >= H_) continue;
#pragma unroll
            for (int kx = 0; kx < 3; ++kx) {
                int xx = wo - 1 + kx;
                if (xx < 0 || xx >= W_) continue;
                acc += xc[y * W_ + xx] * wc[ky * 3 + kx];
            }
        }
    }
    om[idx] = acc;
}

// -------- Stage 2: weight fp32 -> bf16 (layout [o][c*9+k] is already flat) -
__global__ void __launch_bounds__(256)
convert_w_kernel(const float* __restrict__ dw, unsigned short* __restrict__ wbf) {
    int i = blockIdx.x * blockDim.x + threadIdx.x;
    if (i < OC_ * KDIM) wbf[i] = f2bf(dw[i]);
}

// -------- Stage 3: deformable bilinear sampling -> bf16 im2col rows -------
// sT[(n*HW_+p)][c*9+k], row stride KDIM  (B-fragment friendly: contiguous K)
__global__ void __launch_bounds__(256)
deform_sample_kernel(const float* __restrict__ x, const float* __restrict__ om,
                     unsigned short* __restrict__ sT) {
    int idx = blockIdx.x * blockDim.x + threadIdx.x;
    if (idx >= N_ * HW_ * K2_) return;
    int k = idx % 9;
    int p = (idx / 9) & (HW_ - 1);
    int n = idx / (9 * HW_);
    int wo = p & 63, ho = p >> 6;

    const float* omb = om + (size_t)n * 27 * HW_ + p;          // chan stride HW_
    float offy = omb[(2 * k) * HW_];
    float offx = omb[(2 * k + 1) * HW_];
    float m    = 1.0f / (1.0f + __expf(-omb[(18 + k) * HW_])); // sigmoid mask

    float h = (float)(ho - 1 + k / 3) + offy;
    float w = (float)(wo - 1 + k % 3) + offx;
    float y0f = floorf(h), x0f = floorf(w);
    float lh = h - y0f, lw = w - x0f;
    int y0 = (int)y0f, x0 = (int)x0f;

    float w00 = (1.f - lh) * (1.f - lw) * m;
    float w01 = (1.f - lh) * lw * m;
    float w10 = lh * (1.f - lw) * m;
    float w11 = lh * lw * m;

    auto corner = [&](int yi, int xi, float& wgt) -> int {
        if (yi < 0 || yi >= H_ || xi < 0 || xi >= W_) wgt = 0.f;
        int yc = yi < 0 ? 0 : (yi > H_ - 1 ? H_ - 1 : yi);
        int xc = xi < 0 ? 0 : (xi > W_ - 1 ? W_ - 1 : xi);
        return yc * W_ + xc;
    };
    int i00 = corner(y0,     x0,     w00);
    int i01 = corner(y0,     x0 + 1, w01);
    int i10 = corner(y0 + 1, x0,     w10);
    int i11 = corner(y0 + 1, x0 + 1, w11);

    const float* xb = x + (size_t)n * C_ * HW_;
    unsigned short* out = sT + (size_t)(n * HW_ + p) * KDIM + k;
    for (int c = 0; c < C_; ++c) {
        const float* xc = xb + c * HW_;
        float v = w00 * xc[i00] + w01 * xc[i01] + w10 * xc[i10] + w11 * xc[i11];
        out[c * 9] = f2bf(v);
    }
}

// -------- Stage 4: GEMM  D[o][col] = sum_k W[o][k] * S[col][k] ------------
// M=256, K=2304, Ncols=16384.
// Block = 8 waves covering M=256 x N=64. Wave w owns a 32x64 output tile
// (2 M-subtiles x 4 N-subtiles = 8 wmma per K-step). The 32(K)x64(N) B tile
// is staged into LDS with global_load_async_to_lds_b128, double-buffered so
// the DMA of tile kk+32 overlaps the 8 WMMAs on tile kk.
__global__ void __launch_bounds__(256)
dcn_gemm_kernel(const unsigned short* __restrict__ WbfU,
                const unsigned short* __restrict__ sTU,
                float* __restrict__ out) {
    // [buf][col 0..63][k 0..31] bf16 -> fragment-major: lane reads 32B row seg
    __shared__ __align__(16) unsigned short smemB[2][64][32];   // 8 KB

    const bf16* Wbf = (const bf16*)WbfU;
    const int tid  = threadIdx.x;
    const int lane = tid & 31;
    const int wv   = tid >> 5;
    const int row  = lane & 15;      // A row / B column within a 16-wide subtile
    const int half = lane >> 4;
    const int m0   = wv * 32;
    const int col0 = blockIdx.x * 64;

    // Async copy roles: thread t moves one 16B segment of one column.
    const int cidx = tid >> 2;       // 0..63: column of the B tile
    const int seg  = tid & 3;        // 0..3 : 16B segment within 64B K-row
    const unsigned ldsBase =
        (unsigned)(unsigned long long)(uintptr_t)&smemB[0][cidx][seg * 8];
    const unsigned short* gBase = sTU + (size_t)(col0 + cidx) * KDIM + seg * 8;

    // A: lanes 0-15 hold K {0..7,16..23}, lanes 16-31 hold K {8..15,24..31}
    const bf16* aBase0 = Wbf + (size_t)(m0 + row) * KDIM + half * 8;
    const bf16* aBase1 = aBase0 + (size_t)16 * KDIM;

    v8f acc[2][4];
#pragma unroll
    for (int mi = 0; mi < 2; ++mi)
#pragma unroll
        for (int ni = 0; ni < 4; ++ni) acc[mi][ni] = (v8f){};

    // Prologue: DMA tile kk=0 into buf 0.
    asm volatile("global_load_async_to_lds_b128 %0, %1, off"
                 :: "v"(ldsBase), "v"(gBase) : "memory");
    asm volatile("s_wait_asynccnt 0x0" ::: "memory");
    __syncthreads();

    int buf = 0;
    for (int kk = 0; kk < KDIM; kk += 32, buf ^= 1) {
        const int nxt = kk + 32;
        if (nxt < KDIM) {   // prefetch next B tile into the other buffer
            unsigned d = ldsBase + (unsigned)((buf ^ 1) * 64 * 32 * 2);
            const unsigned short* g = gBase + nxt;
            asm volatile("global_load_async_to_lds_b128 %0, %1, off"
                         :: "v"(d), "v"(g) : "memory");
        }

        // B fragments from LDS (K {8h..8h+7, 16+8h..16+8h+7} per lane -> ds_load_b128 x2)
        v16bf B[4];
#pragma unroll
        for (int ni = 0; ni < 4; ++ni)
            B[ni] = __builtin_shufflevector(*(const v8bf*)&smemB[buf][ni * 16 + row][half * 8], *(const v8bf*)&smemB[buf][ni * 16 + row][16 + half * 8], 0, 1, 2, 3, 4, 5, 6, 7, 8, 9, 10, 11, 12, 13, 14, 15);

        // A fragments from global (L2-resident weights)
        v8bf a00 = *(const v8bf*)(aBase0 + kk);
        v8bf a01 = *(const v8bf*)(aBase0 + kk + 16);
        v16bf A0 = __builtin_shufflevector(a00, a01, 0, 1, 2, 3, 4, 5, 6, 7,
                                           8, 9, 10, 11, 12, 13, 14, 15);
        v8bf a10 = *(const v8bf*)(aBase1 + kk);
        v8bf a11 = *(const v8bf*)(aBase1 + kk + 16);
        v16bf A1 = __builtin_shufflevector(a10, a11, 0, 1, 2, 3, 4, 5, 6, 7,
                                           8, 9, 10, 11, 12, 13, 14, 15);

#pragma unroll
        for (int ni = 0; ni < 4; ++ni) {
            acc[0][ni] = __builtin_amdgcn_wmma_f32_16x16x32_bf16(
                false, A0, false, B[ni], (short)0, acc[0][ni], false, false);
            acc[1][ni] = __builtin_amdgcn_wmma_f32_16x16x32_bf16(
                false, A1, false, B[ni], (short)0, acc[1][ni], false, false);
        }

        // Drain this iteration's DMA, then make buffers safe to rotate.
        asm volatile("s_wait_asynccnt 0x0" ::: "memory");
        __syncthreads();
    }

    // C/D layout: VGPR i -> M = 8*(lane>>4)+i, N = lane&15
#pragma unroll
    for (int mi = 0; mi < 2; ++mi) {
        const int obase = m0 + mi * 16 + 8 * half;
#pragma unroll
        for (int ni = 0; ni < 4; ++ni) {
            const int col = col0 + ni * 16 + row;
            const int n = col >> 12;            // col / 4096
            const int p = col & (HW_ - 1);
            float* ob = out + (size_t)n * OC_ * HW_ + p;
#pragma unroll
            for (int i = 0; i < 8; ++i)
                ob[(size_t)(obase + i) * HW_] = acc[mi][ni][i];
        }
    }
}

extern "C" void kernel_launch(void* const* d_in, const int* in_sizes, int n_in,
                              void* d_out, int out_size, void* d_ws, size_t ws_size,
                              hipStream_t stream) {
    const float* x  = (const float*)d_in[0];   // (4,256,64,64)
    const float* ow = (const float*)d_in[1];   // (27,256,3,3)
    const float* ob = (const float*)d_in[2];   // (27,)
    const float* dw = (const float*)d_in[3];   // (256,256,3,3)
    float* out = (float*)d_out;                // (4,256,64,64)

    char* ws = (char*)d_ws;
    float*          om  = (float*)ws;                                // 1,769,472 B
    unsigned short* wbf = (unsigned short*)(ws + 1769472);           // 1,179,648 B
    unsigned short* sT  = (unsigned short*)(ws + 1769472 + 1179648); // 75,497,472 B

    offset_conv_kernel  <<<(N_ * 27 * HW_ + 255) / 256, 256, 0, stream>>>(x, ow, ob, om);
    convert_w_kernel    <<<(OC_ * KDIM + 255) / 256,    256, 0, stream>>>(dw, wbf);
    deform_sample_kernel<<<(N_ * HW_ * K2_ + 255) / 256, 256, 0, stream>>>(x, om, sT);

    dcn_gemm_kernel<<<NCOL / 64, 256, 0, stream>>>(wbf, sT, out);
}
